// CenterAttention3D_45397804319241
// MI455X (gfx1250) — matrix operand order repaired and checked by reading, not compiled, not run on hardware
//
#include <hip/hip_runtime.h>
#include <hip/hip_bf16.h>

// ---------------------------------------------------------------------------
// CenterAttention3D for MI455X (gfx1250, wave32, WMMA + async global->LDS)
//   B=1, N=24^3=13824, C=128, 4 heads x 32 dims, 3x3x3 neighborhood (27)
// GEMMs: v_wmma_f32_16x16x32_bf16 with bf16 hi/lo split (Ahi*Bhi + Ahi*Blo +
// Alo*Bhi) for ~fp32 accuracy. Tiles staged in FRAGMENT-MAJOR LDS layout so
// each lane's 16-bf16 WMMA operand is one 32B contiguous chunk (2x
// ds_load_b128). Conversions work on K-pairs -> packed v2bf ds_store_b32.
// A tile staged via global_load_async_to_lds (ASYNCcnt). Attention stage is
// exact fp32 VALU with wave32 butterfly reductions. Problem is L2/bandwidth
// bound (~2.2 GFLOP over ~50 MB), so the 3x WMMA precision split is free.
// ---------------------------------------------------------------------------

typedef __bf16 bf16_t;
typedef __attribute__((ext_vector_type(2)))  __bf16 v2bf;
typedef __attribute__((ext_vector_type(16))) __bf16 v16bf;
typedef __attribute__((ext_vector_type(8)))  float  v8f;
typedef __attribute__((address_space(3)))    float  lds_f32;

#define CDIM   128
#define DSIDE  24
#define NVOX   (DSIDE * DSIDE * DSIDE)   // 13824
#define NHEADS 4
#define HDIM   32

// Split two fp32 into bf16 hi/lo pairs and store packed (one b32 per array).
__device__ __forceinline__ void split2_store(bf16_t* hiArr, bf16_t* loArr,
                                             int dst, float a, float b) {
  const bf16_t ha = (bf16_t)a, hb = (bf16_t)b;
  const bf16_t la = (bf16_t)(a - (float)ha);
  const bf16_t lb = (bf16_t)(b - (float)hb);
  const v2bf ph = {ha, hb};
  const v2bf pl = {la, lb};
  *(v2bf*)&hiArr[dst] = ph;   // dst is even -> 4B aligned
  *(v2bf*)&loArr[dst] = pl;
}

// ---------------------------------------------------------------------------
// Generic GEMM: out[M x ncols] = in[M x 128] @ W[128 x ncols] + bias
//   grid.x = M/16 (row tiles), grid.y = ncols/128 (column blocks)
//   block  = 256 threads = 8 waves; wave w owns the 16x16 tile at cols w*16
// Fragment-major LDS layouts (ISA 7.12.2, wave32):
//   A (16x32 bf16): lane = M + 16*hiH, element j; K = j+8*(j>=8)+8*hiH
//   B (32x16 bf16): lane = col(l16) + 16*hiH, element j; K = j+8*(j>=8)+8*hiH
// ---------------------------------------------------------------------------
__global__ __launch_bounds__(256) void ca3d_gemm_bf16x3_kernel(
    const float* __restrict__ in, const float* __restrict__ W,
    const float* __restrict__ bias, float* __restrict__ out, int ncols)
{
  __shared__ __align__(16) float  sAf32[16 * 128];           // 8 KB raw A tile
  __shared__ __align__(32) bf16_t sAhi[4 * 32 * 16];         // 4 KB frag-major
  __shared__ __align__(32) bf16_t sAlo[4 * 32 * 16];         // 4 KB
  __shared__ __align__(32) bf16_t sBhi[128 * 2 * 16];        // 8 KB frag-major
  __shared__ __align__(32) bf16_t sBlo[128 * 2 * 16];        // 8 KB

  const int tid  = threadIdx.x;
  const int row0 = blockIdx.x * 16;
  const int col0 = blockIdx.y * 128;

  // ---- Stage A tile (16x128 fp32) via async global->LDS (ASYNCcnt path) ----
  const float* gA = in + (size_t)row0 * CDIM;
  #pragma unroll
  for (int t = 0; t < 8; ++t) {
    const int idx = tid + t * 256;
    const unsigned voff = (unsigned)idx * 4u;
    lds_f32* ldst = (lds_f32*)&sAf32[idx];
    asm volatile("global_load_async_to_lds_b32 %0, %1, %2 offset:0"
                 :: "v"(ldst), "v"(voff), "s"(gA) : "memory");
  }
  asm volatile("s_wait_asynccnt 0x0" ::: "memory");
  __syncthreads();

  // ---- Convert A to frag-major bf16 hi/lo, two K at a time ----
  #pragma unroll
  for (int t = 0; t < 4; ++t) {
    const int p = tid + t * 256;       // pair index, 0..1023
    const int r = p >> 6;              // M row 0..15
    const int k = (p & 63) * 2;        // even K 0..126
    const float2 v2 = *(const float2*)&sAf32[r * 128 + k];
    const int ks  = k >> 5;            // k-step 0..3
    const int kk  = k & 31;
    const int hiH = (kk >> 3) & 1;     // same for kk and kk+1 (kk even)
    const int j0  = (kk & 7) + ((kk & 16) ? 8 : 0);   // even
    const int dst = ((ks * 32 + (r + 16 * hiH)) * 16) + j0;
    split2_store(sAhi, sAlo, dst, v2.x, v2.y);
  }
  __syncthreads();

  const int wave     = tid >> 5;
  const int lane     = tid & 31;
  const int hiHalf   = lane >> 4;
  const int l16      = lane & 15;
  const int colLocal = wave * 16 + l16;

  // Hoist all A fragments (k-loop invariant): 2x ds_load_b128 per fragment.
  v16bf a_hi[4], a_lo[4];
  #pragma unroll
  for (int ks = 0; ks < 4; ++ks) {
    a_hi[ks] = *(const v16bf*)&sAhi[(ks * 32 + lane) * 16];
    a_lo[ks] = *(const v16bf*)&sAlo[(ks * 32 + lane) * 16];
  }

  v8f acc = {};

  #pragma unroll
  for (int ks = 0; ks < 4; ++ks) {
    __syncthreads();   // previous B chunk fully consumed
    // Stage B chunk (32x128 fp32): K-pair x 4-column items; two b128 global
    // loads per item, packed v2bf stores into frag-major layout.
    #pragma unroll
    for (int t = 0; t < 2; ++t) {
      const int item = tid + t * 256;          // 512 items
      const int kk   = (item >> 5) * 2;        // even K within chunk, 0..30
      const int cc4  = (item & 31) * 4;        // column group
      const float4 r0 =
          *(const float4*)&W[(size_t)(ks * 32 + kk) * ncols + (col0 + cc4)];
      const float4 r1 =
          *(const float4*)&W[(size_t)(ks * 32 + kk + 1) * ncols + (col0 + cc4)];
      const int hiH = (kk >> 3) & 1;           // same for kk, kk+1
      const int j0  = (kk & 7) + ((kk & 16) ? 8 : 0);   // even
      const float a0[4] = {r0.x, r0.y, r0.z, r0.w};
      const float a1[4] = {r1.x, r1.y, r1.z, r1.w};
      #pragma unroll
      for (int u = 0; u < 4; ++u) {
        const int dst = (((cc4 + u) * 2 + hiH) * 16) + j0;
        split2_store(sBhi, sBlo, dst, a0[u], a1[u]);
      }
    }
    __syncthreads();

    const v16bf bhi = *(const v16bf*)&sBhi[(colLocal * 2 + hiHalf) * 16];
    const v16bf blo = *(const v16bf*)&sBlo[(colLocal * 2 + hiHalf) * 16];

    acc = __builtin_amdgcn_wmma_f32_16x16x32_bf16(
        false, a_hi[ks], false, bhi, (short)0, acc, false, false);
    acc = __builtin_amdgcn_wmma_f32_16x16x32_bf16(
        false, a_hi[ks], false, blo, (short)0, acc, false, false);
    acc = __builtin_amdgcn_wmma_f32_16x16x32_bf16(
        false, a_lo[ks], false, bhi, (short)0, acc, false, false);
  }

  // Epilogue: C/D layout -> N = lane%16, M = r + 8*(lane>=16)
  const int   col  = col0 + colLocal;
  const float bcol = bias[col];
  #pragma unroll
  for (int r = 0; r < 8; ++r) {
    const int row = row0 + r + hiHalf * 8;
    out[(size_t)row * ncols + col] = acc[r] + bcol;
  }
}

// ---------------------------------------------------------------------------
// Attention: one block (128 thr = 4 waves) per voxel; wave = head, lane = chan.
// Zero-padded neighbors contribute logit == 0 (exp(0) stays in the softmax
// denominator, v contribution is 0), exactly matching the reference's jnp.pad.
// ---------------------------------------------------------------------------
__global__ __launch_bounds__(128) void ca3d_attn_kernel(
    const float* __restrict__ qbuf, const float* __restrict__ kvbuf,
    float* __restrict__ outa)
{
  const int n    = blockIdx.x;
  const int head = threadIdx.x >> 5;
  const int c    = threadIdx.x & 31;

  const int w = n % DSIDE;
  const int h = (n / DSIDE) % DSIDE;
  const int d = n / (DSIDE * DSIDE);

  const float scale = 0.17677669529663688f;   // 32^-0.5
  const float qv = qbuf[(size_t)n * CDIM + head * HDIM + c] * scale;

  float logits[27];
  #pragma unroll
  for (int idx = 0; idx < 27; ++idx) {
    const int nd = d + idx / 9 - 1;
    const int nh = h + (idx / 3) % 3 - 1;
    const int nw = w + idx % 3 - 1;
    float prod = 0.0f;
    if (nd >= 0 && nd < DSIDE && nh >= 0 && nh < DSIDE && nw >= 0 && nw < DSIDE) {
      const int nb = (nd * DSIDE + nh) * DSIDE + nw;
      prod = qv * kvbuf[(size_t)nb * 256 + head * HDIM + c];   // k part
    }
    float r = prod;                     // wave32 butterfly all-reduce
    #pragma unroll
    for (int off = 16; off > 0; off >>= 1) r += __shfl_xor(r, off, 32);
    logits[idx] = r;                    // OOB -> exactly 0, as in reference
  }

  float m = logits[0];
  #pragma unroll
  for (int idx = 1; idx < 27; ++idx) m = fmaxf(m, logits[idx]);

  float denom = 0.0f, accv = 0.0f;
  #pragma unroll
  for (int idx = 0; idx < 27; ++idx) {
    const int nd = d + idx / 9 - 1;
    const int nh = h + (idx / 3) % 3 - 1;
    const int nw = w + idx % 3 - 1;
    const float e = expf(logits[idx] - m);
    denom += e;
    if (nd >= 0 && nd < DSIDE && nh >= 0 && nh < DSIDE && nw >= 0 && nw < DSIDE) {
      const int nb = (nd * DSIDE + nh) * DSIDE + nw;
      accv += e * kvbuf[(size_t)nb * 256 + 128 + head * HDIM + c];  // v part
    }
  }
  outa[(size_t)n * CDIM + head * HDIM + c] = accv / denom;
}

// ---------------------------------------------------------------------------
extern "C" void kernel_launch(void* const* d_in, const int* in_sizes, int n_in,
                              void* d_out, int out_size, void* d_ws, size_t ws_size,
                              hipStream_t stream) {
  const float* x   = (const float*)d_in[0];
  const float* Wq  = (const float*)d_in[1];
  const float* bq  = (const float*)d_in[2];
  const float* Wkv = (const float*)d_in[3];
  const float* bkv = (const float*)d_in[4];
  const float* Wp  = (const float*)d_in[5];
  const float* bp  = (const float*)d_in[6];
  float* out = (float*)d_out;

  // Workspace: q [N,128] | kv [N,256] (k cols 0..127, v cols 128..255) | attn [N,128]
  float* qbuf  = (float*)d_ws;
  float* kvbuf = qbuf  + (size_t)NVOX * CDIM;
  float* abuf  = kvbuf + (size_t)NVOX * 2 * CDIM;

  const dim3 gblk(256);
  const int  mt = NVOX / 16;   // 864 row tiles

  // 1) q = x @ Wq + bq           (scale folded into attention stage)
  ca3d_gemm_bf16x3_kernel<<<dim3(mt, 1), gblk, 0, stream>>>(x, Wq, bq, qbuf, 128);
  // 2) kv = x @ Wkv + bkv        (N x 256)
  ca3d_gemm_bf16x3_kernel<<<dim3(mt, 2), gblk, 0, stream>>>(x, Wkv, bkv, kvbuf, 256);
  // 3) local 27-neighbor softmax attention
  ca3d_attn_kernel<<<dim3(NVOX), dim3(128), 0, stream>>>(qbuf, kvbuf, abuf);
  // 4) out = attn @ Wp + bp
  ca3d_gemm_bf16x3_kernel<<<dim3(mt, 1), gblk, 0, stream>>>(abuf, Wp, bp, out, 128);
}
